// DFINESegTransformer_40467181862959
// MI455X (gfx1250) — matrix operand order repaired and checked by reading, not compiled, not run on hardware
//
#include <hip/hip_runtime.h>
#include <hip/hip_bf16.h>
#include <math.h>

// Problem constants
#define Bsz   4
#define Cch   256
#define Hh    80
#define Ww    80
#define HWp   6400          // 80*80
#define NPIX  (Bsz*HWp)     // 25600
#define Nq    300
#define IDim  128
#define HIDm  512
#define NQall (Bsz*Nq)      // 1200
#define Hup   160
#define Wup   160

typedef __hip_bfloat16 bf16;
typedef __attribute__((ext_vector_type(16))) __bf16 v16bf;
typedef __attribute__((ext_vector_type(8)))  float  v8f;

union Frag { uint4 u[2]; v16bf v; };

// A fragment (16x32 bf16): lane l16 = row m, half = lane>>4.
// element e -> k = kb + 16*(e>>3) + 8*half + (e&7)  => two 16B loads at +0 and +16 elems
__device__ __forceinline__ v16bf load_frag_a(const bf16* p /* row + kb + 8*half */) {
  Frag f;
  f.u[0] = *(const uint4*)(p);
  f.u[1] = *(const uint4*)(p + 16);
  return f.v;
}
// B fragment (32x16 bf16, stored as N rows x K contiguous): lane l16 = col n.
// element e -> k = kb + 16*(e>>3) + 8*half + (e&7)  => two 16B loads at +0 and +16 elems
__device__ __forceinline__ v16bf load_frag_b(const bf16* p /* row + kb + 8*half */) {
  Frag f;
  f.u[0] = *(const uint4*)(p);
  f.u[1] = *(const uint4*)(p + 16);
  return f.v;
}

// ---------------- elementwise / prep kernels ----------------

__global__ void k_f32_to_bf16(const float* __restrict__ in, bf16* __restrict__ out, int n) {
  int i = blockIdx.x * 256 + threadIdx.x;
  if (i < n) out[i] = __float2bfloat16(in[i]);
}

// spatial [B][C][HW] f32 -> pixel-major [B][HW][C] f32
__global__ void k_chw_to_hwc(const float* __restrict__ in, float* __restrict__ out) {
  long i = (long)blockIdx.x * 256 + threadIdx.x;
  if (i >= (long)Bsz * Cch * HWp) return;
  int hw = (int)(i % HWp);
  long t = i / HWp;
  int c = (int)(t % Cch);
  int b = (int)(t / Cch);
  out[((long)b * HWp + hw) * Cch + c] = in[i];
}

// depthwise 3x3 + bias on pixel-major f32: x[B][HW][C] -> out[B][HW][C]
__global__ void k_dwconv(const float* __restrict__ x, const float* __restrict__ w9,
                         const float* __restrict__ bias, float* __restrict__ out) {
  long i = (long)blockIdx.x * 256 + threadIdx.x;
  if (i >= (long)NPIX * Cch) return;
  int c = (int)(i & (Cch - 1));
  long pix = i >> 8;
  int b  = (int)(pix / HWp);
  int hw = (int)(pix % HWp);
  int y = hw / Ww, xx = hw % Ww;
  float acc = bias[c];
  #pragma unroll
  for (int ky = -1; ky <= 1; ky++) {
    int iy = y + ky;
    if (iy < 0 || iy >= Hh) continue;
    #pragma unroll
    for (int kx = -1; kx <= 1; kx++) {
      int ix = xx + kx;
      if (ix < 0 || ix >= Ww) continue;
      acc += w9[c * 9 + (ky + 1) * 3 + (kx + 1)] *
             x[((long)b * HWp + iy * Ww + ix) * Cch + c];
    }
  }
  out[i] = acc;
}

// channel LayerNorm per pixel: dw f32 [pix][256] -> xn bf16 [pix][256]
__global__ void k_ln(const float* __restrict__ dw, const float* __restrict__ g,
                     const float* __restrict__ bt, bf16* __restrict__ out) {
  int p = blockIdx.x * 256 + threadIdx.x;
  if (p >= NPIX) return;
  const float4* row = (const float4*)(dw + (long)p * Cch);
  float s = 0.f, ss = 0.f;
  for (int i = 0; i < Cch / 4; i++) {
    float4 v = row[i];
    s  += v.x + v.y + v.z + v.w;
    ss += v.x * v.x + v.y * v.y + v.z * v.z + v.w * v.w;
  }
  float mean = s * (1.f / Cch);
  float var  = ss * (1.f / Cch) - mean * mean;
  float rstd = rsqrtf(var + 1e-6f);
  bf16* o = out + (long)p * Cch;
  for (int i = 0; i < Cch / 4; i++) {
    float4 v = row[i];
    int c = i * 4;
    o[c + 0] = __float2bfloat16((v.x - mean) * rstd * g[c + 0] + bt[c + 0]);
    o[c + 1] = __float2bfloat16((v.y - mean) * rstd * g[c + 1] + bt[c + 1]);
    o[c + 2] = __float2bfloat16((v.z - mean) * rstd * g[c + 2] + bt[c + 2]);
    o[c + 3] = __float2bfloat16((v.w - mean) * rstd * g[c + 3] + bt[c + 3]);
  }
}

// ---------------- WMMA NT GEMM, 64(M)x16(N) per wave ----------------
// Out[n][m] = sum_k A[m][k]*B[n][k]; M must be a multiple of 64.
// MODE 0: out bf16 = acc + bias[m]
// MODE 1: out bf16 = relu(acc + bias[m])
// MODE 2: pointwise-conv epilogue: o = gelu(acc + bias[m]) + resid[n][m]; store f32(opt)+bf16
// MODE 3: mask: out f32 = acc + bias[0]  (scalar head bias), store-guarded on n
template <int MODE>
__global__ void k_gemm(const bf16* __restrict__ A, const bf16* __restrict__ Bm,
                       int K, int Mt4 /* M/64 */, int Nt, int nValid,
                       const float* __restrict__ bias,
                       const float* __restrict__ resid,
                       float* __restrict__ outF, bf16* __restrict__ outH,
                       int ldo, long sA, long sB, long sC) {
  int wv = threadIdx.x >> 5;
  int lane = threadIdx.x & 31;
  int tile = blockIdx.x * (blockDim.x >> 5) + wv;
  if (tile >= Mt4 * Nt) return;           // wave-uniform branch: EXEC stays all-1s
  int z = blockIdx.y;
  A  += (long)z * sA;
  Bm += (long)z * sB;
  int mt = tile % Mt4, nt = tile / Mt4;
  int half = lane >> 4, l16 = lane & 15;
  int mBase = mt * 64;
  int nRow = nt * 16 + l16;
  if (nRow >= nValid) nRow = nValid - 1;  // clamp (no divergence on loads)
  const bf16* Ap = A  + (long)(mBase + l16) * K + 8  * half;  // + s*16*K per subtile
  const bf16* Bp = Bm + (long)nRow * K + 8 * half;
  const long  As = (long)16 * K;          // subtile row stride

  v8f acc[4];
  #pragma unroll
  for (int s = 0; s < 4; s++) acc[s] = (v8f){0.f, 0.f, 0.f, 0.f, 0.f, 0.f, 0.f, 0.f};

  for (int kb = 0; kb < K; kb += 32) {
    __builtin_prefetch(Bp + kb + 64, 0, 3);
    v16bf b = load_frag_b(Bp + kb);       // one B fragment feeds 4 WMMAs
    #pragma unroll
    for (int s = 0; s < 4; s++) {
      v16bf a = load_frag_a(Ap + As * s + kb);
      acc[s] = __builtin_amdgcn_wmma_f32_16x16x32_bf16(
          /*neg_a=*/false, a, /*neg_b=*/false, b,
          /*c_mod=*/(short)0, acc[s], /*reuse_a=*/false, /*reuse_b=*/false);
    }
  }

  int nOut = nt * 16 + l16;
  bool nOk = nOut < nValid;
  long cBase = (long)z * sC;
  #pragma unroll
  for (int s = 0; s < 4; s++) {
    #pragma unroll
    for (int r = 0; r < 8; r++) {
      int m = mBase + s * 16 + 8 * half + r;
      float v = acc[s][r];
      if (MODE == 0 || MODE == 1) {
        v += bias[m];
        if (MODE == 1) v = fmaxf(v, 0.f);
        if (nOk) outH[cBase + (long)nOut * ldo + m] = __float2bfloat16(v);
      } else if (MODE == 2) {
        v += bias[m];
        float gl = 0.5f * v * (1.f + erff(v * 0.70710678118654752f));  // exact GELU
        float o = gl + resid[(long)nOut * ldo + m];
        if (outF) outF[(long)nOut * ldo + m] = o;
        outH[(long)nOut * ldo + m] = __float2bfloat16(o);
      } else {  // MODE 3
        v += bias[0];
        if (nOk) outF[cBase + (long)nOut * ldo + m] = v;
      }
    }
  }
}

// bilinear x2 upsample (half-pixel centers), per-mask: lo [B*Nq][80][80] -> out [B*Nq][160][160]
__global__ void k_upsample(const float* __restrict__ lo, float* __restrict__ out) {
  long i = (long)blockIdx.x * 256 + threadIdx.x;
  if (i >= (long)Bsz * Nq * Hup * Wup) return;
  int ox = (int)(i % Wup);
  long t = i / Wup;
  int oy = (int)(t % Hup);
  long bq = t / Hup;
  float fy = oy * 0.5f - 0.25f;
  float fx = ox * 0.5f - 0.25f;
  int y0 = (int)floorf(fy); float wy = fy - y0;
  int x0 = (int)floorf(fx); float wx = fx - x0;
  int y0c = min(max(y0, 0), Hh - 1), y1c = min(max(y0 + 1, 0), Hh - 1);
  int x0c = min(max(x0, 0), Ww - 1), x1c = min(max(x0 + 1, 0), Ww - 1);
  const float* src = lo + bq * HWp;
  float v00 = src[y0c * Ww + x0c], v01 = src[y0c * Ww + x1c];
  float v10 = src[y1c * Ww + x0c], v11 = src[y1c * Ww + x1c];
  float v0 = v00 + wx * (v01 - v00);
  float v1 = v10 + wx * (v11 - v10);
  out[i] = v0 + wy * (v1 - v0);
}

// ---------------- host launcher ----------------

extern "C" void kernel_launch(void* const* d_in, const int* in_sizes, int n_in,
                              void* d_out, int out_size, void* d_ws, size_t ws_size,
                              hipStream_t stream) {
  const float* spatial   = (const float*)d_in[0];
  const float* qf        = (const float*)d_in[1];
  const float* dw_w      = (const float*)d_in[2];
  const float* dw_b      = (const float*)d_in[3];
  const float* ln_w      = (const float*)d_in[4];
  const float* ln_b      = (const float*)d_in[5];
  const float* pw_w      = (const float*)d_in[6];
  const float* pw_b      = (const float*)d_in[7];
  const float* sproj_w   = (const float*)d_in[8];
  const float* sproj_b   = (const float*)d_in[9];
  const float* mlp_w1    = (const float*)d_in[10];
  const float* mlp_b1    = (const float*)d_in[11];
  const float* mlp_w2    = (const float*)d_in[12];
  const float* mlp_b2    = (const float*)d_in[13];
  const float* mlp_w3    = (const float*)d_in[14];
  const float* mlp_b3    = (const float*)d_in[15];
  const float* qproj_w   = (const float*)d_in[16];
  const float* qproj_b   = (const float*)d_in[17];
  const float* head_bias = (const float*)d_in[18];
  float* out = (float*)d_out;

  char* w = (char*)d_ws;
  auto alloc = [&](size_t bytes) {
    char* p = w;
    w += (bytes + 255) & ~(size_t)255;
    return p;
  };
  float* XA  = (float*)alloc((size_t)NPIX * Cch * 4);   // residual stream (pixel-major f32)
  float* XB  = (float*)alloc((size_t)NPIX * Cch * 4);   // block output f32 (contiguous after XA)
  float* DW  = (float*)alloc((size_t)NPIX * Cch * 4);   // dwconv scratch
  bf16*  XN  = (bf16*) alloc((size_t)NPIX * Cch * 2);   // LN output bf16
  bf16*  XBH = (bf16*) alloc((size_t)NPIX * Cch * 2);   // block output bf16
  bf16*  SP  = (bf16*) alloc((size_t)NPIX * IDim * 2);  // sproj output bf16
  bf16*  WPW = (bf16*) alloc((size_t)2 * Cch * Cch * 2);
  bf16*  WS  = (bf16*) alloc((size_t)IDim * Cch * 2);
  bf16*  WM1 = (bf16*) alloc((size_t)HIDm * Cch * 2);
  bf16*  WM2 = (bf16*) alloc((size_t)HIDm * HIDm * 2);
  bf16*  WM3 = (bf16*) alloc((size_t)Cch * HIDm * 2);
  bf16*  WQ  = (bf16*) alloc((size_t)IDim * Cch * 2);
  bf16*  QB  = (bf16*) alloc((size_t)NQall * Cch * 2);
  bf16*  H1  = (bf16*) alloc((size_t)NQall * HIDm * 2);
  bf16*  H2  = (bf16*) alloc((size_t)NQall * HIDm * 2);
  bf16*  Q3  = (bf16*) alloc((size_t)NQall * Cch * 2);
  bf16*  QP  = (bf16*) alloc((size_t)NQall * IDim * 2);
  // logits at 80x80 overlay XA+XB (both dead by the time the mask GEMM runs):
  float* LOG80 = XA;  // needs 4*300*6400*4 = 30.7MB <= 52.4MB of XA+XB

  auto cvt = [&](const float* src, bf16* dst, int n) {
    k_f32_to_bf16<<<(n + 255) / 256, 256, 0, stream>>>(src, dst, n);
  };
  // weight / input conversions to bf16
  cvt(pw_w,    WPW, 2 * Cch * Cch);
  cvt(sproj_w, WS,  IDim * Cch);
  cvt(mlp_w1,  WM1, HIDm * Cch);
  cvt(mlp_w2,  WM2, HIDm * HIDm);
  cvt(mlp_w3,  WM3, Cch * HIDm);
  cvt(qproj_w, WQ,  IDim * Cch);
  cvt(qf,      QB,  NQall * Cch);

  // NCHW -> pixel-major
  {
    long n = (long)Bsz * Cch * HWp;
    k_chw_to_hwc<<<(unsigned)((n + 255) / 256), 256, 0, stream>>>(spatial, XA);
  }

  const unsigned dwBlocks = (unsigned)(((long)NPIX * Cch + 255) / 256);
  const unsigned lnBlocks = (NPIX + 255) / 256;
  auto gblocks = [](int tiles) { return (unsigned)((tiles + 7) / 8); };

  // Block 1: dwconv -> LN -> WMMA pointwise + GELU + residual
  k_dwconv<<<dwBlocks, 256, 0, stream>>>(XA, dw_w, dw_b, DW);
  k_ln<<<lnBlocks, 256, 0, stream>>>(DW, ln_w, ln_b, XN);
  k_gemm<2><<<dim3(gblocks(4 * 1600), 1), 256, 0, stream>>>(
      WPW, XN, Cch, 4, 1600, NPIX, pw_b, XA, XB, XBH, Cch, 0, 0, 0);
  // Block 2
  k_dwconv<<<dwBlocks, 256, 0, stream>>>(XB, dw_w + Cch * 9, dw_b + Cch, DW);
  k_ln<<<lnBlocks, 256, 0, stream>>>(DW, ln_w + Cch, ln_b + Cch, XN);
  k_gemm<2><<<dim3(gblocks(4 * 1600), 1), 256, 0, stream>>>(
      WPW + Cch * Cch, XN, Cch, 4, 1600, NPIX, pw_b + Cch, XB, nullptr, XBH, Cch, 0, 0, 0);

  // sproj at 80x80 (commutes with the linear upsample): SP[pix][128]
  k_gemm<0><<<dim3(gblocks(2 * 1600), 1), 256, 0, stream>>>(
      WS, XBH, Cch, 2, 1600, NPIX, sproj_b, nullptr, nullptr, SP, IDim, 0, 0, 0);

  // Query MLP chain (flattened over batch: 1200 rows)
  k_gemm<1><<<dim3(gblocks(8 * 75), 1), 256, 0, stream>>>(
      WM1, QB, Cch, 8, 75, NQall, mlp_b1, nullptr, nullptr, H1, HIDm, 0, 0, 0);
  k_gemm<1><<<dim3(gblocks(8 * 75), 1), 256, 0, stream>>>(
      WM2, H1, HIDm, 8, 75, NQall, mlp_b2, nullptr, nullptr, H2, HIDm, 0, 0, 0);
  k_gemm<0><<<dim3(gblocks(4 * 75), 1), 256, 0, stream>>>(
      WM3, H2, HIDm, 4, 75, NQall, mlp_b3, nullptr, nullptr, Q3, Cch, 0, 0, 0);
  k_gemm<0><<<dim3(gblocks(2 * 75), 1), 256, 0, stream>>>(
      WQ, Q3, Cch, 2, 75, NQall, qproj_b, nullptr, nullptr, QP, IDim, 0, 0, 0);

  // Mask GEMM at 80x80: A = SP (per-batch 6400x128), B = QP (per-batch 300x128, clamped)
  // Out f32 [b][query][6400] = acc + head_bias
  k_gemm<3><<<dim3(gblocks(100 * 19), Bsz), 256, 0, stream>>>(
      SP, QP, IDim, 100, 19, Nq, head_bias, nullptr, LOG80, nullptr, HWp,
      (long)HWp * IDim, (long)Nq * IDim, (long)Nq * HWp);

  // Bilinear x2 upsample of the 300 mask maps -> final [4,300,160,160]
  {
    long n = (long)Bsz * Nq * Hup * Wup;
    k_upsample<<<(unsigned)((n + 255) / 256), 256, 0, stream>>>(LOG80, out);
  }
}
